// GIN_49168785604940
// MI455X (gfx1250) — matrix operand order repaired and checked by reading, not compiled, not run on hardware
//
#include <hip/hip_runtime.h>
#include <math.h>

// ---------------------------------------------------------------------------
// GIN + SAGPool forward for gfx1250 (MI455X).
// GEMMs: v_wmma_f32_16x16x32_f16 (f16 A/B, f32 accumulate).
// A tiles DMA'd into LDS with global_load_async_to_lds_b128 (ASYNCcnt),
// weights pre-swizzled into the WMMA B-operand lane layout.
// ---------------------------------------------------------------------------

typedef _Float16 v16h __attribute__((ext_vector_type(16)));
typedef float    v8f  __attribute__((ext_vector_type(8)));

#define D 128
#define BGR 32
#define EDGES 1048576
#define BN_EPS 1e-5f

// Async DMA 16 bytes: global -> LDS (per active lane). lds_byte is the
// wave-relative LDS byte address (low 32 bits of the generic LDS pointer).
__device__ __forceinline__ void async_copy16(unsigned lds_byte, const void* gptr)
{
    asm volatile("global_load_async_to_lds_b128 %0, %1, off"
                 :: "v"(lds_byte), "v"(gptr)
                 : "memory");
}

__device__ __forceinline__ void wait_async0()
{
    asm volatile("s_wait_asynccnt 0x0" ::: "memory");
}

// ---------------------------------------------------------------------------
// Pack a 128x128 row-major f32 weight (W[k][c]) into a contiguous-k WMMA B-operand
// layout as f16 pairs: dword index = ((ct*4 + kb)*32 + lane)*8 + v, where per
// lane: half = lane>>4, n = lane&15, K = kb*32 + 2v + 16*half, and the dword
// packs { W[K][ct*16+n], W[K+1][ct*16+n] } (low half = even K).
// ---------------------------------------------------------------------------
__global__ void pack_weights(const float* __restrict__ W, unsigned* __restrict__ Wp)
{
    int tid = blockIdx.x * blockDim.x + threadIdx.x;   // 8192 total
    if (tid >= 8192) return;
    int v    = tid & 7;
    int lane = (tid >> 3) & 31;
    int kb   = (tid >> 8) & 3;
    int ct   = tid >> 10;
    int half = lane >> 4;
    int n    = lane & 15;
    int kk   = kb * 32 + 2 * v + 16 * half;
    int c    = ct * 16 + n;
    union { _Float16 f[2]; unsigned u; } pk;
    pk.f[0] = (_Float16)W[(size_t)kk * D + c];
    pk.f[1] = (_Float16)W[(size_t)(kk + 1) * D + c];
    Wp[tid] = pk.u;
}

// ---------------------------------------------------------------------------
// Y[nrows x 128] = epilogue( (A (+A2)) @ W + bias ),  W pre-packed (Wp).
// Block = 256 threads = 8 waves -> one 16-row tile; wave w -> cols [16w,16w+16).
// A (and A2) tiles DMA'd to LDS asynchronously; GIN add done on LDS->reg path.
// ---------------------------------------------------------------------------
template <bool HAS_A2, bool HAS_BN, bool RELU>
__global__ __launch_bounds__(256)
void linear_wmma(const float* __restrict__ A, const float* __restrict__ A2,
                 const unsigned* __restrict__ Wp, const float* __restrict__ bias,
                 const float* __restrict__ bng, const float* __restrict__ bnb,
                 const float* __restrict__ bnm, const float* __restrict__ bnv,
                 float* __restrict__ Y)
{
    __shared__ float at[16 * D];                 // 8 KB staged A tile
    __shared__ float at2[16 * D];                // 8 KB staged A2 tile (GIN agg)
    const int tid  = threadIdx.x;
    const int row0 = blockIdx.x << 4;

    // async DMA staging: 2 chunks of 16B per thread per tile, fully coalesced
    {
        const unsigned ldsA = (unsigned)(size_t)(const void*)at;
        const char*    ga   = (const char*)(A + (size_t)row0 * D);
        const unsigned b0   = (unsigned)tid * 16u;
        async_copy16(ldsA + b0,         ga + b0);
        async_copy16(ldsA + b0 + 4096u, ga + b0 + 4096u);
        if (HAS_A2) {
            const unsigned ldsB = (unsigned)(size_t)(const void*)at2;
            const char*    gb   = (const char*)(A2 + (size_t)row0 * D);
            async_copy16(ldsB + b0,         gb + b0);
            async_copy16(ldsB + b0 + 4096u, gb + b0 + 4096u);
        }
        wait_async0();           // drain this wave's ASYNCcnt
    }
    __syncthreads();             // publish LDS to all 8 waves

    const int lane = tid & 31;
    const int wave = tid >> 5;
    const int half = lane >> 4;
    const int lm   = lane & 15;

    v8f acc = {};
#pragma unroll
    for (int kb = 0; kb < 4; ++kb) {
        v16h a = {};
#pragma unroll
        for (int v = 0; v < 8; ++v) {
            // A in contiguous k order (same as Wp): VGPR v -> K pair 2v; +16 for lanes 16-31
            int kk = kb * 32 + 2 * v + 16 * half;
            float2 f = *(const float2*)(at + lm * D + kk);   // ds_load_b64
            if (HAS_A2) {
                float2 f2 = *(const float2*)(at2 + lm * D + kk);
                f.x += f2.x; f.y += f2.y;
            }
            a[2 * v]     = (_Float16)f.x;
            a[2 * v + 1] = (_Float16)f.y;
        }
        v16h b = *(const v16h*)(Wp + ((((wave << 2) + kb) << 5) + lane) * 8);
        acc = __builtin_amdgcn_wmma_f32_16x16x32_f16(
                  false, a, false, b, (short)0, acc, false, false);
    }

    const int c = (wave << 4) + lm;
    const float biasv = bias[c];
    float scale = 1.f, shift = 0.f;
    if (HAS_BN) {
        const float sc = bng[c] * rsqrtf(bnv[c] + BN_EPS);
        scale = sc;
        shift = bnb[c] - bnm[c] * sc;
    }
#pragma unroll
    for (int r = 0; r < 8; ++r) {
        float val = acc[r] + biasv;
        if (HAS_BN) val = val * scale + shift;
        if (RELU)   val = fmaxf(val, 0.f);
        Y[(size_t)(row0 + r + 8 * half) * D + c] = val;
    }
}

// ---------------------------------------------------------------------------
// agg[dst[e]] += h[src[e]] * mask[e]   (32 lanes per edge, float4 + atomics)
// ---------------------------------------------------------------------------
__global__ __launch_bounds__(256)
void aggregate_k(const float* __restrict__ h, const int* __restrict__ src,
                 const int* __restrict__ dst, const float* __restrict__ mask,
                 float* __restrict__ agg, int E)
{
    long long tid = (long long)blockIdx.x * blockDim.x + threadIdx.x;
    int e = (int)(tid >> 5);
    int q = (int)(tid & 31);
    if (e >= E) return;
    float m = mask[e];
    if (m == 0.f) return;
    int s = src[e], d = dst[e];
    const float4 v = *(const float4*)(h + (size_t)s * D + q * 4);
    float* o = agg + (size_t)d * D + q * 4;
    atomicAdd(o + 0, v.x * m);
    atomicAdd(o + 1, v.y * m);
    atomicAdd(o + 2, v.z * m);
    atomicAdd(o + 3, v.w * m);
}

// ---------------------------------------------------------------------------
// score[i] = dot(agg[i], Wrel) + dot(h[i], Wroot) + b      (1 wave per node)
// ---------------------------------------------------------------------------
__global__ __launch_bounds__(256)
void score_k(const float* __restrict__ h, const float* __restrict__ agg,
             const float* __restrict__ Wroot, const float* __restrict__ Wrel,
             const float* __restrict__ pb, float* __restrict__ score, int N)
{
    int lane = threadIdx.x & 31;
    int wave = threadIdx.x >> 5;
    int node = blockIdx.x * 8 + wave;
    if (node >= N) return;
    const float* hp = h   + (size_t)node * D;
    const float* ap = agg + (size_t)node * D;
    float s = 0.f;
#pragma unroll
    for (int i = 0; i < 4; ++i) {
        int f = lane + 32 * i;
        s += hp[f] * Wroot[f] + ap[f] * Wrel[f];
    }
#pragma unroll
    for (int off = 16; off > 0; off >>= 1) s += __shfl_xor(s, off, 32);
    if (lane == 0) score[node] = s + pb[0];
}

// ---------------------------------------------------------------------------
// Per-graph top-k (bitonic sort in LDS, descending score, ties -> lower idx,
// matching jax.lax.top_k), then new_h = h[perm] * tanh(score), new_id map.
// One block (1024 threads) per graph; npg <= 2048.
// ---------------------------------------------------------------------------
__global__ __launch_bounds__(1024)
void sag_pool(const float* __restrict__ score, const float* __restrict__ h,
              float* __restrict__ newh, int* __restrict__ new_id, int npg, int k)
{
    __shared__ float sv[2048];
    __shared__ int   si[2048];
    int g   = blockIdx.x;
    int tid = threadIdx.x;
    for (int i = tid; i < 2048; i += 1024) {
        if (i < npg) { sv[i] = score[g * npg + i]; si[i] = i; }
        else         { sv[i] = -3.4e38f;           si[i] = i; }
    }
    __syncthreads();
    for (int ks = 2; ks <= 2048; ks <<= 1) {
        for (int j = ks >> 1; j > 0; j >>= 1) {
            for (int i = tid; i < 2048; i += 1024) {
                int ixj = i ^ j;
                if (ixj > i) {
                    float a = sv[i], b = sv[ixj];
                    int ia = si[i], ib = si[ixj];
                    bool aWorse = (a < b) || (a == b && ia > ib);
                    bool desc = ((i & ks) == 0);
                    if (desc ? aWorse : !aWorse) {
                        sv[i] = b; sv[ixj] = a;
                        si[i] = ib; si[ixj] = ia;
                    }
                }
            }
            __syncthreads();
        }
    }
    for (int j = tid; j < k; j += 1024) {
        new_id[g * npg + si[j]] = g * k + j;
        sv[j] = tanhf(sv[j]);
    }
    __syncthreads();
    for (int t = tid; t < k * D; t += 1024) {
        int j = t >> 7, d = t & 127;
        newh[(size_t)(g * k + j) * D + d] =
            h[(size_t)(g * npg + si[j]) * D + d] * sv[j];
    }
}

__global__ void remap_edges(int* src, int* dst, float* mask,
                            const int* __restrict__ new_id, int E)
{
    int e = blockIdx.x * blockDim.x + threadIdx.x;
    if (e >= E) return;
    float m = mask[e];
    int ns = new_id[src[e]];
    int nd = new_id[dst[e]];
    bool valid = (m > 0.f) && (ns >= 0) && (nd >= 0);
    src[e]  = valid ? ns : 0;
    dst[e]  = valid ? nd : 0;
    mask[e] = valid ? 1.f : 0.f;
}

__global__ void init_edges(const int* __restrict__ s, const int* __restrict__ d,
                           int* ws, int* wd, float* wm, int E)
{
    int e = blockIdx.x * blockDim.x + threadIdx.x;
    if (e >= E) return;
    ws[e] = s[e]; wd[e] = d[e]; wm[e] = 1.f;
}

__global__ void fill_f32(float* p, float v, long long n)
{
    long long i = (long long)blockIdx.x * blockDim.x + threadIdx.x;
    if (i < n) p[i] = v;
}

__global__ void fill_i32(int* p, int v, int n)
{
    int i = blockIdx.x * blockDim.x + threadIdx.x;
    if (i < n) p[i] = v;
}

// out[g] = concat(mean over nodes, max over nodes)  -> [32, 256]
__global__ void readout(const float* __restrict__ h, float* __restrict__ out, int npg)
{
    int g = blockIdx.x, d = threadIdx.x;   // 128 threads
    float sum = 0.f, mx = -3.4e38f;
    const float* p = h + (size_t)g * npg * D + d;
    for (int n = 0; n < npg; ++n) {
        float v = p[(size_t)n * D];
        sum += v;
        mx = fmaxf(mx, v);
    }
    out[g * 256 + d]       = sum / (float)npg;
    out[g * 256 + 128 + d] = mx;
}

// ---------------------------------------------------------------------------
// Host orchestration (graph-capture safe; scratch lives only in d_ws)
// ---------------------------------------------------------------------------
extern "C" void kernel_launch(void* const* d_in, const int* in_sizes, int n_in,
                              void* d_out, int out_size, void* d_ws, size_t ws_size,
                              hipStream_t stream)
{
    (void)in_sizes; (void)n_in; (void)out_size; (void)ws_size;
    const int NPG = 2048, E = EDGES;
    const int N0 = BGR * NPG;            // 65536
    const int K1 = 1639, K2 = 1312;

    // setup_inputs() insertion order
    const float* x      = (const float*)d_in[0];
    const int*   src_in = (const int*)d_in[1];
    const int*   dst_in = (const int*)d_in[2];
    const float* Wenc   = (const float*)d_in[3];
    const float* benc   = (const float*)d_in[4];
    const float *W1[4], *b1[4], *g1[4], *bb1[4], *m1[4], *v1[4];
    const float *W2[4], *b2[4], *g2[4], *bb2[4], *m2[4], *v2[4];
    int idx = 5;
    for (int i = 0; i < 4; ++i) {
        W1[i]  = (const float*)d_in[idx++]; b1[i]  = (const float*)d_in[idx++];
        g1[i]  = (const float*)d_in[idx++]; bb1[i] = (const float*)d_in[idx++];
        m1[i]  = (const float*)d_in[idx++]; v1[i]  = (const float*)d_in[idx++];
        W2[i]  = (const float*)d_in[idx++]; b2[i]  = (const float*)d_in[idx++];
        g2[i]  = (const float*)d_in[idx++]; bb2[i] = (const float*)d_in[idx++];
        m2[i]  = (const float*)d_in[idx++]; v2[i]  = (const float*)d_in[idx++];
    }
    const float* Wroot1 = (const float*)d_in[idx++];
    const float* Wrel1  = (const float*)d_in[idx++];
    const float* pb1    = (const float*)d_in[idx++];
    const float* Wroot2 = (const float*)d_in[idx++];
    const float* Wrel2  = (const float*)d_in[idx++];
    const float* pb2    = (const float*)d_in[idx++];

    // workspace carve-up (~110 MB)
    char* wsb = (char*)d_ws;
    size_t off = 0;
    auto carve = [&](size_t bytes) -> void* {
        void* p = wsb + off;
        off = (off + bytes + 255) & ~(size_t)255;
        return p;
    };
    float*    bufA  = (float*)carve((size_t)N0 * D * 4);
    float*    bufB  = (float*)carve((size_t)N0 * D * 4);
    float*    bufC  = (float*)carve((size_t)N0 * D * 4);
    int*      wsrc  = (int*)carve((size_t)E * 4);
    int*      wdst  = (int*)carve((size_t)E * 4);
    float*    wmask = (float*)carve((size_t)E * 4);
    float*    score = (float*)carve((size_t)N0 * 4);
    int*      newid = (int*)carve((size_t)N0 * 4);
    unsigned* packW = (unsigned*)carve((size_t)9 * 8192 * 4);

    // pre-swizzle all 9 weight matrices into WMMA B layout (f16)
    const float* Wmats[9] = { Wenc, W1[0], W2[0], W1[1], W2[1], W1[2], W2[2], W1[3], W2[3] };
    for (int m = 0; m < 9; ++m)
        pack_weights<<<32, 256, 0, stream>>>(Wmats[m], packW + (size_t)m * 8192);

    init_edges<<<E / 256, 256, 0, stream>>>(src_in, dst_in, wsrc, wdst, wmask, E);

    float* h = bufA; float* t = bufB; float* agg = bufC;

    // encoder: h = x @ Wenc + benc
    linear_wmma<false, false, false><<<N0 / 16, 256, 0, stream>>>(
        x, nullptr, packW, benc, nullptr, nullptr, nullptr, nullptr, h);

    int N = N0, npg = NPG;
    for (int i = 0; i < 4; ++i) {
        long long nn = (long long)N * D;
        fill_f32<<<(unsigned)((nn + 255) / 256), 256, 0, stream>>>(agg, 0.f, nn);
        aggregate_k<<<E / 8, 256, 0, stream>>>(h, wsrc, wdst, wmask, agg, E);
        const unsigned* pW1 = packW + (size_t)(1 + 2 * i) * 8192;
        const unsigned* pW2 = packW + (size_t)(2 + 2 * i) * 8192;
        // t = relu(bn1((h+agg)@W1 + b1))   (GIN z = h + agg fused on LDS path)
        linear_wmma<true, true, true><<<N / 16, 256, 0, stream>>>(
            h, agg, pW1, b1[i], g1[i], bb1[i], m1[i], v1[i], t);
        // h = relu(bn2(t@W2 + b2))
        linear_wmma<false, true, true><<<N / 16, 256, 0, stream>>>(
            t, nullptr, pW2, b2[i], g2[i], bb2[i], m2[i], v2[i], h);
        if (i == 0 || i == 3) {
            int k = (i == 0) ? K1 : K2;
            const float* Wroot = (i == 0) ? Wroot1 : Wroot2;
            const float* Wrel  = (i == 0) ? Wrel1  : Wrel2;
            const float* pb    = (i == 0) ? pb1    : pb2;
            fill_f32<<<(unsigned)((nn + 255) / 256), 256, 0, stream>>>(agg, 0.f, nn);
            aggregate_k<<<E / 8, 256, 0, stream>>>(h, wsrc, wdst, wmask, agg, E);
            score_k<<<(N + 7) / 8, 256, 0, stream>>>(h, agg, Wroot, Wrel, pb, score, N);
            fill_i32<<<(N + 255) / 256, 256, 0, stream>>>(newid, -1, N);
            sag_pool<<<BGR, 1024, 0, stream>>>(score, h, t, newid, npg, k);
            remap_edges<<<E / 256, 256, 0, stream>>>(wsrc, wdst, wmask, newid, E);
            float* tmp = h; h = t; t = tmp;
            npg = k; N = BGR * k;
        }
    }
    readout<<<BGR, 128, 0, stream>>>(h, (float*)d_out, npg);
}
